// MultiHeadAttention_47175920780042
// MI455X (gfx1250) — compile-verified, matrix operand order checked by reading, not run on hardware
//
#include <hip/hip_runtime.h>
#include <stdint.h>

// ---------------------------------------------------------------------------
// MHA forward for MI455X (gfx1250, wave32, WMMA bf16 16x16x32 f32-accumulate)
// + TDM async K-tile staging, double-buffered LDS, ping-pong-pipelined GEMMs.
// __launch_bounds__(256,1) keeps the 32-row-per-wave GEMM tiles in VGPRs.
// ---------------------------------------------------------------------------

typedef __attribute__((ext_vector_type(16))) __bf16 v16bf;
typedef __attribute__((ext_vector_type(8)))  float  v8f;
typedef unsigned int u32;
typedef __attribute__((ext_vector_type(4))) u32 v4u;
typedef __attribute__((ext_vector_type(8))) int  v8i_t;
typedef __attribute__((ext_vector_type(4))) int  v4i_t;

#if defined(__has_builtin)
#  if __has_builtin(__builtin_amdgcn_tensor_load_to_lds)
#    define HAVE_TDM 1
#  else
#    define HAVE_TDM 0
#  endif
#else
#  define HAVE_TDM 0
#endif

union Frag16 { uint4 u[2]; v16bf v; };

__device__ __forceinline__ unsigned short f32_to_bf16(float f) {
  unsigned int u = __float_as_uint(f);
  u += 0x7FFFu + ((u >> 16) & 1u);   // round-to-nearest-even
  return (unsigned short)(u >> 16);
}

// 16x32 bf16 WMMA fragment from a row-major [rows x ld] bf16 matrix.
// A-layout (ISA 7.12.2): lane holds row (lane&15); element j holds
// k = k0 + (j&8)*2 + (lane>>4)*8 + (j&7) -> two contiguous 16B loads.
// Works for B too when the matrix is stored [n][k] (transposed).
__device__ __forceinline__ v16bf ld_frag(const unsigned short* __restrict__ base,
                                         int row0, int ld, int k0) {
  int lane = threadIdx.x & 31;
  const unsigned short* p =
      base + (size_t)(row0 + (lane & 15)) * ld + k0 + ((lane >> 4) << 3);
  Frag16 f;
  f.u[0] = *(const uint4*)(p);
  f.u[1] = *(const uint4*)(p + 16);
  return f.v;
}

__device__ __forceinline__ v8f wmma_bf16(v16bf a, v16bf b, v8f c) {
  return __builtin_amdgcn_wmma_f32_16x16x32_bf16(
      false, a, false, b, (short)0, c, false, false);
}

// ------------------------------- prep kernels ------------------------------

__global__ void cvt_kernel(const float* __restrict__ in,
                           unsigned short* __restrict__ out, int n) {
  int i = blockIdx.x * 256 + threadIdx.x;
  if (i < n) out[i] = f32_to_bf16(in[i]);
}

// W[h][1024][64] f32 -> Wt[h][64][1024] bf16 (transpose per head)
__global__ void cvt_w_kernel(const float* __restrict__ in,
                             unsigned short* __restrict__ out) {
  int i = blockIdx.x * 256 + threadIdx.x;          // 16*65536 total
  int h = i >> 16, rem = i & 65535;
  int k = rem >> 6, n = rem & 63;
  out[(h << 16) + (n << 10) + k] = f32_to_bf16(in[i]);
}

// Wo[1024(HV)][1024(dm)] f32 -> Wot[1024(dm)][1024(HV)] bf16
__global__ void cvt_wo_kernel(const float* __restrict__ in,
                              unsigned short* __restrict__ out) {
  int i = blockIdx.x * 256 + threadIdx.x;          // 1048576 total
  int r = i >> 10, c = i & 1023;
  out[(c << 10) + r] = f32_to_bf16(in[i]);
}

// ------------------------- QKV projection (48 GEMMs) -----------------------
// grid: (M/256=16, 1, 48)  z = mat*16 + h ; block 256 (8 waves)
// Wave: 32 rows x 64 cols (8 accum tiles); ping-pong pipelined K-loop
// (no fragment copies -> no mov chains / WMMA hazard NOPs between MMAs).

__global__ void __launch_bounds__(256, 1) qkv_proj_kernel(
    const unsigned short* __restrict__ Xq, const unsigned short* __restrict__ Xk,
    const unsigned short* __restrict__ Xv,
    const unsigned short* __restrict__ Wqt, const unsigned short* __restrict__ Wkt,
    const unsigned short* __restrict__ Wvt,
    const float* __restrict__ bq, const float* __restrict__ bk,
    const float* __restrict__ bv,
    unsigned short* __restrict__ Qb, unsigned short* __restrict__ Kb,
    unsigned short* __restrict__ Vb) {
  int z = blockIdx.z;
  int mat = z >> 4, h = z & 15;
  const unsigned short* X;
  const unsigned short* Wt;
  const float* bias;
  unsigned short* Out;
  if (mat == 0)      { X = Xq; Wt = Wqt; bias = bq; Out = Qb; }
  else if (mat == 1) { X = Xk; Wt = Wkt; bias = bk; Out = Kb; }
  else               { X = Xv; Wt = Wvt; bias = bv; Out = Vb; }
  Wt   += h * 64 * 1024;
  bias += h * 64;
  Out  += (size_t)h * 4096 * 64;

  int wave = threadIdx.x >> 5, lane = threadIdx.x & 31;
  int half = lane >> 4, l15 = lane & 15;
  int m0 = blockIdx.x * 256 + wave * 32;

  v8f acc[8] = {};                       // [rowHalf(2) x ntile(4)]
  v16bf a0[2], a1[2], b[2][4];

  auto load_set = [&](int s, int k0) {
    a0[s] = ld_frag(X, m0,      1024, k0);
    a1[s] = ld_frag(X, m0 + 16, 1024, k0);
#pragma unroll
    for (int nt = 0; nt < 4; nt++) b[s][nt] = ld_frag(Wt, nt * 16, 1024, k0);
  };
  auto compute_set = [&](int s) {
#pragma unroll
    for (int nt = 0; nt < 4; nt++) {
      acc[nt]     = wmma_bf16(a0[s], b[s][nt], acc[nt]);
      acc[4 + nt] = wmma_bf16(a1[s], b[s][nt], acc[4 + nt]);
    }
  };

  load_set(0, 0);
#pragma unroll 1
  for (int k0 = 0; k0 < 960; k0 += 64) {
    load_set(1, k0 + 32);
    compute_set(0);
    load_set(0, k0 + 64);
    compute_set(1);
  }
  load_set(1, 992);
  compute_set(0);
  compute_set(1);

#pragma unroll
  for (int g = 0; g < 2; g++) {
#pragma unroll
    for (int nt = 0; nt < 4; nt++) {
#pragma unroll
      for (int r = 0; r < 8; r++) {
        int row = m0 + g * 16 + r + half * 8;
        int col = nt * 16 + l15;
        Out[(size_t)row * 64 + col] =
            f32_to_bf16(acc[g * 4 + nt][r] + bias[col]);
      }
    }
  }
}

// ---------------------- flash attention (online softmax) -------------------
// grid: (S/128=16, 1, H*B=32); block 256 (8 waves, each owns 16 query rows)
// K tiles staged by the Tensor Data Mover (double-buffered, padded rows);
// V tiles staged transposed by the block; P relaid out via per-wave LDS.

#define KT_LD 72   // 32 x 64 K tile row stride (bf16 units) incl. TDM pad
#define VT_LD 40   // 64 x 32 transposed-V tile, padded
#define PT_LD 40   // per-wave 16 x 32 P scratch, padded

#if HAVE_TDM
__device__ __forceinline__ void tdm_load_k_tile(u32 lds_off,
                                                const unsigned short* gptr) {
  unsigned long long ga = (unsigned long long)(uintptr_t)gptr;
  // D# group0: count=1 (user), lds_addr, global_addr[56:0], type=2
  v4u g0 = { 1u,
             lds_off,
             (u32)(ga & 0xFFFFFFFFu),
             (u32)((ga >> 32) & 0x01FFFFFFu) | (2u << 30) };
  // D# group1: data_size=2B(1), pad_enable, pad_interval=4 (32 DWORDs/row),
  // pad_amount=3 (4 DWORDs = 16B -> LDS row stride 72 bf16);
  // tensor_dim0=64, tensor_dim1=32, tile_dim0=64, tile_dim1=32,
  // tensor_dim0_stride=64.
  v8i_t g1 = { (int)((1u << 16) | (1u << 20) | (4u << 22) | (3u << 25)),
               (int)(64u << 16),      // tensor_dim0[15:0] @ bits 63:48
               (int)(32u << 16),      // tensor_dim0[31:16]=0, tensor_dim1[15:0]
               (int)(64u << 16),      // tensor_dim1[31:16]=0, tile_dim0=64
               (int)32,               // tile_dim1=32, tile_dim2=0
               (int)64,               // tensor_dim0_stride[31:0]
               0, 0 };
  v4i_t z4 = { 0, 0, 0, 0 };
#if __clang_major__ >= 23
  v8i_t z8 = { 0, 0, 0, 0, 0, 0, 0, 0 };
  __builtin_amdgcn_tensor_load_to_lds(g0, g1, z4, z4, z8, 0);
#else
  __builtin_amdgcn_tensor_load_to_lds(g0, g1, z4, z4, 0);
#endif
}
#endif

__global__ void __launch_bounds__(256, 1) attn_kernel(
    const unsigned short* __restrict__ Qb,
    const unsigned short* __restrict__ Kb,
    const unsigned short* __restrict__ Vb,
    unsigned short* __restrict__ Ab) {
  __shared__ __align__(16) unsigned short Kt[2][32 * KT_LD];
  __shared__ __align__(16) unsigned short Vt[2][64 * VT_LD];
  __shared__ __align__(16) unsigned short Pt[8 * 16 * PT_LD];

  const int S = 2048;
  int hb = blockIdx.z;                     // h*B + b
  size_t headoff = (size_t)hb * S * 64;
  const unsigned short* Q = Qb + headoff;
  const unsigned short* K = Kb + headoff;
  const unsigned short* V = Vb + headoff;

  int wave = threadIdx.x >> 5, lane = threadIdx.x & 31;
  int half = lane >> 4, l15 = lane & 15;
  int m0 = blockIdx.x * 128 + wave * 16;

  v16bf qf[2];
  qf[0] = ld_frag(Q, m0, 64, 0);
  qf[1] = ld_frag(Q, m0, 64, 32);

  v8f o[4] = {};
  float mrow[8], lrow[8];
#pragma unroll
  for (int r = 0; r < 8; r++) { mrow[r] = -1e30f; lrow[r] = 0.0f; }

  unsigned short* myP = &Pt[wave * 16 * PT_LD];
  const float scale = 0.125f;              // 1/sqrt(64)

  // cooperative stage of tile t into buffer buf: V transposed (+K if no TDM)
  auto stage_tile = [&](int t, int buf) {
    int row = threadIdx.x >> 3;            // 0..31 (key within tile)
    int c8  = (threadIdx.x & 7) << 3;      // 0,8,..,56
    size_t key = (size_t)t * 32 + row;
    uint4 vq = *(const uint4*)(V + key * 64 + c8);
    const unsigned short* pv = (const unsigned short*)&vq;
#pragma unroll
    for (int i = 0; i < 8; i++) Vt[buf][(c8 + i) * VT_LD + row] = pv[i];
#if !HAVE_TDM
    uint4 kq = *(const uint4*)(K + key * 64 + c8);
    *(uint4*)(&Kt[buf][row * KT_LD + c8]) = kq;
#endif
  };

  const int nT = S / 32;
  // prologue: tile 0 into buffer 0
  stage_tile(0, 0);
#if HAVE_TDM
  if (threadIdx.x < 32) {
    tdm_load_k_tile((u32)(uintptr_t)&Kt[0][0], K);
    __builtin_amdgcn_s_wait_tensorcnt(0);
  }
#endif
  __syncthreads();

#pragma unroll 1
  for (int t = 0; t < nT; t++) {
    int cur = t & 1;
    if (t + 1 < nT) {
      stage_tile(t + 1, cur ^ 1);          // overlaps with compute below
#if HAVE_TDM
      if (threadIdx.x < 32)
        tdm_load_k_tile((u32)(uintptr_t)&Kt[cur ^ 1][0],
                        K + (size_t)(t + 1) * 32 * 64);
#endif
    }

    // scores = Q (16x64) * K^T (64x32) -> two 16x16 f32 tiles
    v8f sc[2] = {};
#pragma unroll
    for (int ks = 0; ks < 2; ks++) {
#pragma unroll
      for (int nt = 0; nt < 2; nt++) {
        Frag16 bf;
        const unsigned short* p =
            &Kt[cur][(nt * 16 + l15) * KT_LD + ks * 32 + (half << 3)];
        bf.u[0] = *(const uint4*)(p);
        bf.u[1] = *(const uint4*)(p + 16);
        sc[nt] = wmma_bf16(qf[ks], bf.v, sc[nt]);
      }
    }

    // online softmax; C-layout rows live across 16-lane halves
    float corr[8];
#pragma unroll
    for (int r = 0; r < 8; r++) {
      float s0 = sc[0][r] * scale;
      float s1 = sc[1][r] * scale;
      float tm = fmaxf(s0, s1);
      tm = fmaxf(tm, __shfl_xor(tm, 1, 32));
      tm = fmaxf(tm, __shfl_xor(tm, 2, 32));
      tm = fmaxf(tm, __shfl_xor(tm, 4, 32));
      tm = fmaxf(tm, __shfl_xor(tm, 8, 32));
      float mn = fmaxf(mrow[r], tm);
      float c  = __expf(mrow[r] - mn);
      float p0 = __expf(s0 - mn);
      float p1 = __expf(s1 - mn);
      float rs = p0 + p1;
      rs += __shfl_xor(rs, 1, 32);
      rs += __shfl_xor(rs, 2, 32);
      rs += __shfl_xor(rs, 4, 32);
      rs += __shfl_xor(rs, 8, 32);
      lrow[r] = lrow[r] * c + rs;
      mrow[r] = mn;
      corr[r] = c;
      int mrl = r + half * 8;
      myP[mrl * PT_LD + l15]      = f32_to_bf16(p0);
      myP[mrl * PT_LD + 16 + l15] = f32_to_bf16(p1);
    }

#pragma unroll
    for (int nt = 0; nt < 4; nt++)
#pragma unroll
      for (int r = 0; r < 8; r++) o[nt][r] = o[nt][r] * corr[r];

    // P (16x32) as A fragment from per-wave LDS scratch
    Frag16 pf;
    {
      const unsigned short* p = &myP[l15 * PT_LD + (half << 3)];
      pf.u[0] = *(const uint4*)(p);
      pf.u[1] = *(const uint4*)(p + 16);
    }
    // O += P * V
#pragma unroll
    for (int nt = 0; nt < 4; nt++) {
      Frag16 vf;
      const unsigned short* p = &Vt[cur][(nt * 16 + l15) * VT_LD + (half << 3)];
      vf.u[0] = *(const uint4*)(p);
      vf.u[1] = *(const uint4*)(p + 16);
      o[nt] = wmma_bf16(pf.v, vf.v, o[nt]);
    }

#if HAVE_TDM
    if (threadIdx.x < 32) __builtin_amdgcn_s_wait_tensorcnt(0);
#endif
    __syncthreads();
  }

  // normalize, store in (h,b,s,v) order (same addressing as Q)
  unsigned short* Aout = Ab + headoff;
#pragma unroll
  for (int r = 0; r < 8; r++) {
    float inv = 1.0f / lrow[r];
    int row = m0 + r + half * 8;
#pragma unroll
    for (int nt = 0; nt < 4; nt++) {
      int col = nt * 16 + l15;
      Aout[(size_t)row * 64 + col] = f32_to_bf16(o[nt][r] * inv);
    }
  }
}

// --------------------------- output projection -----------------------------
// out[4096,1024] f32 = Att[4096,1024]bf16 * Wo + bo (linear read of (h,b,s,v)
// buffer reproduces the reference's head-major cat/reshape exactly).
// grid: (16, 16); block 256 (8 waves); wave = 32 rows x 64 cols, ping-pong.

__global__ void __launch_bounds__(256, 1) out_proj_kernel(
    const unsigned short* __restrict__ Abuf,
    const unsigned short* __restrict__ Wot,
    const float* __restrict__ bo,
    float* __restrict__ Out) {
  int wave = threadIdx.x >> 5, lane = threadIdx.x & 31;
  int half = lane >> 4, l15 = lane & 15;
  int m0 = blockIdx.x * 256 + wave * 32;
  int n0 = blockIdx.y * 64;

  v8f acc[8] = {};
  v16bf a0[2], a1[2], b[2][4];

  auto load_set = [&](int s, int k0) {
    a0[s] = ld_frag(Abuf, m0,      1024, k0);
    a1[s] = ld_frag(Abuf, m0 + 16, 1024, k0);
#pragma unroll
    for (int nt = 0; nt < 4; nt++)
      b[s][nt] = ld_frag(Wot, n0 + nt * 16, 1024, k0);
  };
  auto compute_set = [&](int s) {
#pragma unroll
    for (int nt = 0; nt < 4; nt++) {
      acc[nt]     = wmma_bf16(a0[s], b[s][nt], acc[nt]);
      acc[4 + nt] = wmma_bf16(a1[s], b[s][nt], acc[4 + nt]);
    }
  };

  load_set(0, 0);
#pragma unroll 1
  for (int k0 = 0; k0 < 960; k0 += 64) {
    load_set(1, k0 + 32);
    compute_set(0);
    load_set(0, k0 + 64);
    compute_set(1);
  }
  load_set(1, 992);
  compute_set(0);
  compute_set(1);

#pragma unroll
  for (int g = 0; g < 2; g++) {
#pragma unroll
    for (int nt = 0; nt < 4; nt++) {
#pragma unroll
      for (int r = 0; r < 8; r++) {
        int row = m0 + g * 16 + r + half * 8;
        int col = n0 + nt * 16 + l15;
        Out[(size_t)row * 1024 + col] = acc[g * 4 + nt][r] + bo[col];
      }
    }
  }
}

// ------------------------------- launcher ----------------------------------

extern "C" void kernel_launch(void* const* d_in, const int* in_sizes, int n_in,
                              void* d_out, int out_size, void* d_ws, size_t ws_size,
                              hipStream_t stream) {
  (void)in_sizes; (void)n_in; (void)out_size; (void)ws_size;

  const float* q   = (const float*)d_in[0];
  const float* k_  = (const float*)d_in[1];
  const float* v   = (const float*)d_in[2];
  const float* Wq  = (const float*)d_in[3];
  const float* bq  = (const float*)d_in[4];
  const float* Wk  = (const float*)d_in[5];
  const float* bk  = (const float*)d_in[6];
  const float* Wv  = (const float*)d_in[7];
  const float* bv  = (const float*)d_in[8];
  const float* Wo  = (const float*)d_in[9];
  const float* bo  = (const float*)d_in[10];

  char* w = (char*)d_ws;
  const size_t MB = 1024ull * 1024ull;
  unsigned short* Xq  = (unsigned short*)(w + 0 * MB);   // 4096x1024 bf16
  unsigned short* Xk  = (unsigned short*)(w + 8 * MB);
  unsigned short* Xv  = (unsigned short*)(w + 16 * MB);
  unsigned short* Wqt = (unsigned short*)(w + 24 * MB);  // [16][64][1024] bf16
  unsigned short* Wkt = (unsigned short*)(w + 26 * MB);
  unsigned short* Wvt = (unsigned short*)(w + 28 * MB);
  unsigned short* Wot = (unsigned short*)(w + 30 * MB);  // [1024][1024] bf16
  unsigned short* Qb  = (unsigned short*)(w + 32 * MB);  // [16][4096][64] bf16
  unsigned short* Kb  = (unsigned short*)(w + 40 * MB);
  unsigned short* Vb  = (unsigned short*)(w + 48 * MB);
  unsigned short* Ab  = (unsigned short*)(w + 56 * MB);  // [4096][1024] bf16

  cvt_kernel<<<16384, 256, 0, stream>>>(q,  Xq, 4096 * 1024);
  cvt_kernel<<<16384, 256, 0, stream>>>(k_, Xk, 4096 * 1024);
  cvt_kernel<<<16384, 256, 0, stream>>>(v,  Xv, 4096 * 1024);
  cvt_w_kernel<<<4096, 256, 0, stream>>>(Wq, Wqt);
  cvt_w_kernel<<<4096, 256, 0, stream>>>(Wk, Wkt);
  cvt_w_kernel<<<4096, 256, 0, stream>>>(Wv, Wvt);
  cvt_wo_kernel<<<4096, 256, 0, stream>>>(Wo, Wot);

  qkv_proj_kernel<<<dim3(16, 1, 48), 256, 0, stream>>>(
      Xq, Xk, Xv, Wqt, Wkt, Wvt, bq, bk, bv, Qb, Kb, Vb);

  attn_kernel<<<dim3(16, 1, 32), 256, 0, stream>>>(Qb, Kb, Vb, Ab);

  out_proj_kernel<<<dim3(16, 16), 256, 0, stream>>>(Ab, Wot, bo, (float*)d_out);
}
